// RealFinancialGNN_11931419148387
// MI455X (gfx1250) — compile-verified, matrix operand order checked by reading, not run on hardware
//
#include <hip/hip_runtime.h>

// ---------------------------------------------------------------------------
// Problem constants (from the reference)
// ---------------------------------------------------------------------------
#define NN 10000          // nodes
#define EE 320000         // edges
#define EP 330000         // edges + self loops
#define HD 128            // hidden dim (= F_IN)
#define BNINV 0.9999950000374997f   // 1/sqrt(1+1e-5)

typedef __attribute__((ext_vector_type(16))) _Float16 v16h;
typedef __attribute__((ext_vector_type(8)))  float    v8f;

// ---- ordered-uint encoding for float atomic max ---------------------------
__device__ __forceinline__ unsigned f2ord(float f) {
  unsigned u = __float_as_uint(f);
  return (u & 0x80000000u) ? ~u : (u | 0x80000000u);
}
__device__ __forceinline__ float ord2f(unsigned u) {
  return (u & 0x80000000u) ? __uint_as_float(u & 0x7FFFFFFFu)
                           : __uint_as_float(~u);
}

// ---------------------------------------------------------------------------
// f32 -> f16 conversion (vectorized: float4 in, 4xf16 out) and
// weight transpose (B operand stored column-major f16)
// ---------------------------------------------------------------------------
__global__ void k_f32_to_f16_v4(const float* __restrict__ s,
                                _Float16* __restrict__ d, int n4) {
  int i = blockIdx.x * blockDim.x + threadIdx.x;
  if (i >= n4) return;
  float4 v = ((const float4*)s)[i];
  union { _Float16 h[4]; uint2 u; } o;
  o.h[0] = (_Float16)v.x; o.h[1] = (_Float16)v.y;
  o.h[2] = (_Float16)v.z; o.h[3] = (_Float16)v.w;
  ((uint2*)d)[i] = o.u;
}

__global__ void k_transpose_f16(const float* __restrict__ W,
                                _Float16* __restrict__ Wt) {
  int i = blockIdx.x * blockDim.x + threadIdx.x;   // 128*128
  int k = i >> 7, n = i & 127;
  Wt[n * HD + k] = (_Float16)W[i];
}

// ---------------------------------------------------------------------------
// WMMA GEMM: Out[M,128] = A[M,128] @ W[128,128]
//   block = 256 threads = 8 waves; wave w -> column tile w; block -> 16 rows.
//   MODE 0: store raw accumulator (xp)
//   MODE 1: encoder epilogue  relu((acc+bias)*gamma*BNINV+beta), + f16 copy
// A-fragment layout (16-bit 16x32, ISA 7.12.2): lane L holds row M=L%16;
//   halves 0..7 = K = (L/16)*8 .. +7 ; halves 8..15 = K = 16+(L/16)*8 .. +7
// B pre-transposed (column major) => identical addressing pattern.
// ---------------------------------------------------------------------------
template <int MODE>
__global__ __launch_bounds__(256) void k_gemm_wmma(
    const _Float16* __restrict__ A, const _Float16* __restrict__ Wt,
    float* __restrict__ Out, _Float16* __restrict__ OutH,
    const float* __restrict__ bias, const float* __restrict__ gamma,
    const float* __restrict__ beta)
{
  const int lane    = threadIdx.x & 31;
  const int wave    = threadIdx.x >> 5;     // 0..7 -> column tile
  const int rowBase = blockIdx.x << 4;      // 16 rows per block (10000/16=625)
  const int colBase = wave << 4;
  const int mr      = lane & 15;
  const int hi      = lane >> 4;

  const _Float16* arow = A  + (size_t)(rowBase + mr) * HD;
  const _Float16* brow = Wt + (size_t)(colBase + mr) * HD;

  v8f acc = {};
#pragma unroll
  for (int k0 = 0; k0 < HD; k0 += 32) {
    union { uint4 u[2]; v16h v; } fa, fb;
    fa.u[0] = *(const uint4*)(arow + k0 +      hi * 8);
    fa.u[1] = *(const uint4*)(arow + k0 + 16 + hi * 8);
    fb.u[0] = *(const uint4*)(brow + k0 +      hi * 8);
    fb.u[1] = *(const uint4*)(brow + k0 + 16 + hi * 8);
    acc = __builtin_amdgcn_wmma_f32_16x16x32_f16(false, fa.v, false, fb.v,
                                                 (short)0, acc, false, false);
  }

  const int n = colBase + mr;               // output column
  float bi = 0.f, ga = 0.f, be = 0.f;
  if (MODE == 1) { bi = bias[n]; ga = gamma[n] * BNINV; be = beta[n]; }

#pragma unroll
  for (int r = 0; r < 8; ++r) {
    int m = rowBase + r + 8 * hi;           // C/D layout: lanes 16-31 -> M=8+r
    float v = acc[r];
    if (MODE == 1) {
      v = (v + bi) * ga + be;
      v = v > 0.f ? v : 0.f;
      OutH[(size_t)m * HD + n] = (_Float16)v;
    }
    Out[(size_t)m * HD + n] = v;
  }
}

// ---------------------------------------------------------------------------
// per-node attention coefficients  a_s / a_d  [N,8]
// ---------------------------------------------------------------------------
__global__ void k_attn_sd(const float* __restrict__ xp,
                          const float* __restrict__ att_src,
                          const float* __restrict__ att_dst,
                          float* __restrict__ a_s, float* __restrict__ a_d) {
  int i = blockIdx.x * blockDim.x + threadIdx.x;
  if (i >= NN * 8) return;
  int node = i >> 3, h = i & 7;
  const float* xr = xp + (size_t)node * HD + h * 16;
  float s = 0.f, d = 0.f;
#pragma unroll
  for (int c = 0; c < 16; c += 4) {
    float4 v = *(const float4*)(xr + c);
    float4 as = *(const float4*)(att_src + h * 16 + c);
    float4 ad = *(const float4*)(att_dst + h * 16 + c);
    s += v.x * as.x + v.y * as.y + v.z * as.z + v.w * as.w;
    d += v.x * ad.x + v.y * ad.y + v.z * ad.z + v.w * ad.w;
  }
  a_s[i] = s; a_d[i] = d;
}

// mean of edge_attr over E rows (3 dims -> 3 blocks)
__global__ void k_ea_mean(const float* __restrict__ ea, float* __restrict__ mean) {
  __shared__ float red[256];
  int d = blockIdx.x;
  float s = 0.f;
  for (int e = threadIdx.x; e < EE; e += blockDim.x) s += ea[e * 3 + d];
  red[threadIdx.x] = s; __syncthreads();
  for (int w = 128; w > 0; w >>= 1) {
    if (threadIdx.x < w) red[threadIdx.x] += red[threadIdx.x + w];
    __syncthreads();
  }
  if (threadIdx.x == 0) mean[d] = red[0] / (float)EE;
}

// wae[d,h] = sum_c we[d, h*16+c] * att_edge[h,c]   (3x8 pre-contraction)
__global__ void k_wae(const float* __restrict__ we, const float* __restrict__ att_e,
                      float* __restrict__ wae) {
  int i = threadIdx.x;
  if (i >= 24) return;
  int d = i >> 3, h = i & 7;
  float s = 0.f;
#pragma unroll
  for (int c = 0; c < 16; ++c) s += we[d * HD + h * 16 + c] * att_e[h * 16 + c];
  wae[d * 8 + h] = s;
}

// a_e[e,h] = ea[e] . wae[:,h]   (self-loop rows use the mean attr)
__global__ void k_a_e(const float* __restrict__ ea, const float* __restrict__ mean,
                      const float* __restrict__ wae, float* __restrict__ a_e) {
  int i = blockIdx.x * blockDim.x + threadIdx.x;
  if (i >= EP * 8) return;
  int e = i >> 3, h = i & 7;
  float d0, d1, d2;
  if (e < EE) { d0 = ea[e * 3]; d1 = ea[e * 3 + 1]; d2 = ea[e * 3 + 2]; }
  else        { d0 = mean[0];   d1 = mean[1];       d2 = mean[2]; }
  a_e[i] = d0 * wae[h] + d1 * wae[8 + h] + d2 * wae[16 + h];
}

// zero agg / den, init segment max to -inf (ordered encoding)
__global__ void k_seg_init(float* __restrict__ agg, float* __restrict__ den,
                           unsigned* __restrict__ mx) {
  int i = blockIdx.x * blockDim.x + threadIdx.x;
  if (i < NN * HD) agg[i] = 0.f;
  if (i < NN * 8)  { den[i] = 0.f; mx[i] = 0x007FFFFFu; }   // f2ord(-inf)
}

// logits + segment max (per dst,head)
__global__ void k_logit(const int* __restrict__ ei, const float* __restrict__ a_s,
                        const float* __restrict__ a_d, const float* __restrict__ a_e,
                        float* __restrict__ logit, unsigned* __restrict__ mx) {
  int i = blockIdx.x * blockDim.x + threadIdx.x;
  if (i >= EP * 8) return;
  int e = i >> 3, h = i & 7;
  int s = (e < EE) ? ei[e]      : (e - EE);
  int d = (e < EE) ? ei[EE + e] : (e - EE);
  float l = a_s[s * 8 + h] + a_d[d * 8 + h] + a_e[i];
  l = (l > 0.f) ? l : 0.2f * l;                 // leaky_relu(0.2)
  logit[i] = l;
  atomicMax(&mx[d * 8 + h], f2ord(l));
}

// num = exp(logit - max[dst]);  den[dst] += num   (num overwrites logit)
__global__ void k_expnum(const int* __restrict__ ei, float* __restrict__ num,
                         const unsigned* __restrict__ mx, float* __restrict__ den) {
  int i = blockIdx.x * blockDim.x + threadIdx.x;
  if (i >= EP * 8) return;
  int e = i >> 3, h = i & 7;
  int d = (e < EE) ? ei[EE + e] : (e - EE);
  float nu = __expf(num[i] - ord2f(mx[d * 8 + h]));
  num[i] = nu;
  atomicAdd(&den[d * 8 + h], nu);
}

// agg[dst, c] += (num/den[dst]) * xp[src, c]
// one thread per (edge, 4-channel group): indices + alpha loaded once,
// xp gather is a single b128 load, 4 f32 atomics out.
__global__ void k_scatter(const int* __restrict__ ei, const float* __restrict__ num,
                          const float* __restrict__ den, const float* __restrict__ xp,
                          float* __restrict__ agg) {
  int i = blockIdx.x * blockDim.x + threadIdx.x;   // EP*32 = 10.56M
  if (i >= EP * 32) return;
  int e = i >> 5, q = i & 31;       // q: which group of 4 channels
  int h = q >> 2;                   // head = (q*4)/16
  int s = (e < EE) ? ei[e]      : (e - EE);
  int d = (e < EE) ? ei[EE + e] : (e - EE);
  float alpha = num[e * 8 + h] / den[d * 8 + h];
  float4 xv = *(const float4*)(xp + (size_t)s * HD + q * 4);
  float* ag = agg + (size_t)d * HD + q * 4;
  atomicAdd(ag + 0, alpha * xv.x);
  atomicAdd(ag + 1, alpha * xv.y);
  atomicAdd(ag + 2, alpha * xv.z);
  atomicAdd(ag + 3, alpha * xv.w);
}

// bias + BN + relu + 0.5*residual; h updated in place; f16 copy for next GEMM
// vectorized over 4 channels.
__global__ void k_combine(const float* __restrict__ agg, const float* __restrict__ bias,
                          const float* __restrict__ g, const float* __restrict__ b,
                          float* __restrict__ h, _Float16* __restrict__ hb) {
  int i = blockIdx.x * blockDim.x + threadIdx.x;   // NN*HD/4
  if (i >= NN * HD / 4) return;
  int c4 = i & 31;                                 // channel group
  float4 av = ((const float4*)agg)[i];
  float4 bv = ((const float4*)bias)[c4];
  float4 gv = ((const float4*)g)[c4];
  float4 ev = ((const float4*)b)[c4];
  float4 hv = ((const float4*)h)[i];
  float4 o;
  o.x = (av.x + bv.x) * (gv.x * BNINV) + ev.x;
  o.y = (av.y + bv.y) * (gv.y * BNINV) + ev.y;
  o.z = (av.z + bv.z) * (gv.z * BNINV) + ev.z;
  o.w = (av.w + bv.w) * (gv.w * BNINV) + ev.w;
  o.x = (o.x > 0.f ? o.x : 0.f) + 0.5f * hv.x;
  o.y = (o.y > 0.f ? o.y : 0.f) + 0.5f * hv.y;
  o.z = (o.z > 0.f ? o.z : 0.f) + 0.5f * hv.z;
  o.w = (o.w > 0.f ? o.w : 0.f) + 0.5f * hv.w;
  ((float4*)h)[i] = o;
  union { _Float16 hh[4]; uint2 u; } ov;
  ov.hh[0] = (_Float16)o.x; ov.hh[1] = (_Float16)o.y;
  ov.hh[2] = (_Float16)o.z; ov.hh[3] = (_Float16)o.w;
  ((uint2*)hb)[i] = ov.u;
}

// global mean/max pool -> pooled[256]
// one block per 4-channel group; threads stride rows with b128 loads.
__global__ void k_pool(const float* __restrict__ h, float* __restrict__ pooled) {
  __shared__ float4 ssum[256], smax[256];
  int q = blockIdx.x;                              // 0..31
  float4 s = {0.f, 0.f, 0.f, 0.f};
  float4 m = {-3.4e38f, -3.4e38f, -3.4e38f, -3.4e38f};
  for (int i = threadIdx.x; i < NN; i += blockDim.x) {
    float4 v = *(const float4*)(h + (size_t)i * HD + q * 4);
    s.x += v.x; s.y += v.y; s.z += v.z; s.w += v.w;
    m.x = fmaxf(m.x, v.x); m.y = fmaxf(m.y, v.y);
    m.z = fmaxf(m.z, v.z); m.w = fmaxf(m.w, v.w);
  }
  ssum[threadIdx.x] = s; smax[threadIdx.x] = m; __syncthreads();
  for (int w = 128; w > 0; w >>= 1) {
    if (threadIdx.x < w) {
      float4 a = ssum[threadIdx.x], bq = ssum[threadIdx.x + w];
      a.x += bq.x; a.y += bq.y; a.z += bq.z; a.w += bq.w;
      ssum[threadIdx.x] = a;
      float4 c = smax[threadIdx.x], dq = smax[threadIdx.x + w];
      c.x = fmaxf(c.x, dq.x); c.y = fmaxf(c.y, dq.y);
      c.z = fmaxf(c.z, dq.z); c.w = fmaxf(c.w, dq.w);
      smax[threadIdx.x] = c;
    }
    __syncthreads();
  }
  if (threadIdx.x == 0) {
    float4 a = ssum[0], c = smax[0];
    pooled[q * 4 + 0] = a.x / (float)NN; pooled[q * 4 + 1] = a.y / (float)NN;
    pooled[q * 4 + 2] = a.z / (float)NN; pooled[q * 4 + 3] = a.w / (float)NN;
    pooled[HD + q * 4 + 0] = c.x; pooled[HD + q * 4 + 1] = c.y;
    pooled[HD + q * 4 + 2] = c.z; pooled[HD + q * 4 + 3] = c.w;
  }
}

// tiny head MLPs, single block
__global__ void k_heads(const float* __restrict__ pooled,
                        const float* rw0, const float* rb0, const float* rw1,
                        const float* rb1, const float* rw2, const float* rb2,
                        const float* vw0, const float* vb0, const float* vw1, const float* vb1,
                        const float* tw0, const float* tb0, const float* tw1, const float* tb1,
                        float* __restrict__ out) {
  __shared__ float zin[256], z1[128], z2[64];
  int t = threadIdx.x;
  zin[t] = pooled[t];
  __syncthreads();
  // risk: 256 -> 128 relu -> 64 relu -> 1
  if (t < 128) {
    float s = rb0[t];
    for (int k = 0; k < 256; ++k) s += zin[k] * rw0[k * 128 + t];
    z1[t] = s > 0.f ? s : 0.f;
  }
  __syncthreads();
  if (t < 64) {
    float s = rb1[t];
    for (int k = 0; k < 128; ++k) s += z1[k] * rw1[k * 64 + t];
    z2[t] = s > 0.f ? s : 0.f;
  }
  __syncthreads();
  if (t == 0) {
    float s = rb2[0];
    for (int k = 0; k < 64; ++k) s += z2[k] * rw2[k];
    out[0] = s;
  }
  __syncthreads();
  // vol: 256 -> 64 relu -> 1, softplus
  if (t < 64) {
    float s = vb0[t];
    for (int k = 0; k < 256; ++k) s += zin[k] * vw0[k * 64 + t];
    z2[t] = s > 0.f ? s : 0.f;
  }
  __syncthreads();
  if (t == 0) {
    float s = vb1[0];
    for (int k = 0; k < 64; ++k) s += z2[k] * vw1[k];
    out[1] = (s > 20.f) ? s : log1pf(__expf(s));
  }
  __syncthreads();
  // ret: 256 -> 64 relu -> 1
  if (t < 64) {
    float s = tb0[t];
    for (int k = 0; k < 256; ++k) s += zin[k] * tw0[k * 64 + t];
    z2[t] = s > 0.f ? s : 0.f;
  }
  __syncthreads();
  if (t == 0) {
    float s = tb1[0];
    for (int k = 0; k < 64; ++k) s += z2[k] * tw1[k];
    out[2] = s;
  }
}

// ---------------------------------------------------------------------------
// Host orchestration
// Input order: x, edge_index, edge_attr, then params flattened as a JAX
// pytree (dict keys alphabetical): bn_b[2], bn_g[2], enc_b, enc_be, enc_g,
// enc_w, gat[0]{att_dst,att_edge,att_src,bias,w,we}, gat[1]{...},
// ret[w0,b0,w1,b1], risk[w0,b0,w1,b1,w2,b2], vol[w0,b0,w1,b1]
// d_out: [risk, vol, ret, h(10000x128)] flat f32.
// ---------------------------------------------------------------------------
extern "C" void kernel_launch(void* const* d_in, const int* in_sizes, int n_in,
                              void* d_out, int out_size, void* d_ws, size_t ws_size,
                              hipStream_t stream) {
  (void)in_sizes; (void)n_in; (void)out_size; (void)ws_size;

  const float* x   = (const float*)d_in[0];
  const int*   ei  = (const int*)  d_in[1];
  const float* ea  = (const float*)d_in[2];
  const float* bn_b[2]     = {(const float*)d_in[3],  (const float*)d_in[4]};
  const float* bn_g[2]     = {(const float*)d_in[5],  (const float*)d_in[6]};
  const float* enc_b       =  (const float*)d_in[7];
  const float* enc_be      =  (const float*)d_in[8];
  const float* enc_g       =  (const float*)d_in[9];
  const float* enc_w       =  (const float*)d_in[10];
  const float* att_dst[2]  = {(const float*)d_in[11], (const float*)d_in[17]};
  const float* att_edge[2] = {(const float*)d_in[12], (const float*)d_in[18]};
  const float* att_src[2]  = {(const float*)d_in[13], (const float*)d_in[19]};
  const float* gbias[2]    = {(const float*)d_in[14], (const float*)d_in[20]};
  const float* gw[2]       = {(const float*)d_in[15], (const float*)d_in[21]};
  const float* gwe[2]      = {(const float*)d_in[16], (const float*)d_in[22]};
  const float* ret_p[4];  for (int i = 0; i < 4; ++i) ret_p[i]  = (const float*)d_in[23 + i];
  const float* risk_p[6]; for (int i = 0; i < 6; ++i) risk_p[i] = (const float*)d_in[27 + i];
  const float* vol_p[4];  for (int i = 0; i < 4; ++i) vol_p[i]  = (const float*)d_in[33 + i];

  // workspace carve (256B aligned)
  char* wsp = (char*)d_ws;
  size_t off = 0;
  auto take = [&](size_t bytes) -> void* {
    off = (off + 255) & ~(size_t)255;
    void* p = wsp + off;
    off += bytes;
    return p;
  };
  _Float16* hbA = (_Float16*)take((size_t)NN * HD * 2);
  _Float16* hbB = (_Float16*)take((size_t)NN * HD * 2);
  _Float16* wt  = (_Float16*)take((size_t)HD * HD * 2);
  float* xp     = (float*)take((size_t)NN * HD * 4);
  float* h      = (float*)take((size_t)NN * HD * 4);
  float* agg    = (float*)take((size_t)NN * HD * 4);
  float* a_s    = (float*)take((size_t)NN * 8 * 4);
  float* a_d    = (float*)take((size_t)NN * 8 * 4);
  float* a_e    = (float*)take((size_t)EP * 8 * 4);
  float* lg     = (float*)take((size_t)EP * 8 * 4);      // logits, reused as num
  unsigned* mx  = (unsigned*)take((size_t)NN * 8 * 4);
  float* den    = (float*)take((size_t)NN * 8 * 4);
  float* mean3  = (float*)take(3 * 4);
  float* wae    = (float*)take(24 * 4);
  float* pooled = (float*)take(256 * 4);

  const dim3 B(256);
  const int gNH  = (NN * HD + 255) / 256;       // per-element grid
  const int gNH4 = (NN * HD / 4 + 255) / 256;   // per-float4 grid
  const int gN8  = (NN * 8 + 255) / 256;
  const int gE8  = (EP * 8 + 255) / 256;
  const int gE32 = (EP * 32 + 255) / 256;

  // encoder: h = relu(BN(x @ enc_w + enc_b))
  k_f32_to_f16_v4<<<gNH4, B, 0, stream>>>(x, hbA, NN * HD / 4);
  k_transpose_f16<<<64, B, 0, stream>>>(enc_w, wt);
  k_gemm_wmma<1><<<NN / 16, B, 0, stream>>>(hbA, wt, h, hbB, enc_b, enc_g, enc_be);
  k_ea_mean<<<3, B, 0, stream>>>(ea, mean3);

  _Float16* hin = hbB; _Float16* hout = hbA;
  for (int l = 0; l < 2; ++l) {
    k_transpose_f16<<<64, B, 0, stream>>>(gw[l], wt);
    k_gemm_wmma<0><<<NN / 16, B, 0, stream>>>(hin, wt, xp, nullptr, nullptr, nullptr, nullptr);
    k_attn_sd<<<gN8, B, 0, stream>>>(xp, att_src[l], att_dst[l], a_s, a_d);
    k_wae<<<1, 32, 0, stream>>>(gwe[l], att_edge[l], wae);
    k_a_e<<<gE8, B, 0, stream>>>(ea, mean3, wae, a_e);
    k_seg_init<<<gNH, B, 0, stream>>>(agg, den, mx);
    k_logit<<<gE8, B, 0, stream>>>(ei, a_s, a_d, a_e, lg, mx);
    k_expnum<<<gE8, B, 0, stream>>>(ei, lg, mx, den);
    k_scatter<<<gE32, B, 0, stream>>>(ei, lg, den, xp, agg);
    k_combine<<<gNH4, B, 0, stream>>>(agg, gbias[l], bn_g[l], bn_b[l], h, hout);
    _Float16* tmp = hin; hin = hout; hout = tmp;
  }

  k_pool<<<32, B, 0, stream>>>(h, pooled);
  float* out = (float*)d_out;
  k_heads<<<1, B, 0, stream>>>(pooled,
      risk_p[0], risk_p[1], risk_p[2], risk_p[3], risk_p[4], risk_p[5],
      vol_p[0], vol_p[1], vol_p[2], vol_p[3],
      ret_p[0], ret_p[1], ret_p[2], ret_p[3],
      out);
  hipMemcpyAsync(out + 3, h, (size_t)NN * HD * 4, hipMemcpyDeviceToDevice, stream);
}
